// PatientGraphModel_85177791414326
// MI455X (gfx1250) — compile-verified, matrix operand order checked by reading, not run on hardware
//
#include <hip/hip_runtime.h>
#include <hip/hip_bf16.h>

typedef __bf16 bf16_t;
typedef __attribute__((ext_vector_type(16))) __bf16 bf16x16;
typedef __attribute__((ext_vector_type(8)))  float  floatx8;

// ---------------------------------------------------------------------------
// Small elementwise / graph kernels
// ---------------------------------------------------------------------------
__global__ void k_fill(float* __restrict__ p, float v, long long n) {
    long long i = (long long)blockIdx.x * blockDim.x + threadIdx.x;
    if (i < n) p[i] = v;
}

__global__ void k_deg_accum(const long long* __restrict__ dst, float* __restrict__ deg,
                            long long E) {
    long long e = (long long)blockIdx.x * blockDim.x + threadIdx.x;
    if (e < E) atomicAdd(&deg[dst[e]], 1.0f);
}

__global__ void k_rsqrt_inplace(float* __restrict__ p, long long n) {
    long long i = (long long)blockIdx.x * blockDim.x + threadIdx.x;
    if (i < n) p[i] = rsqrtf(p[i]);   // deg >= 1 always (self loop)
}

// out[i,f] = h[i,f] * dinv[i]^2 + bias[f]   (self-loop message + bias, full overwrite)
template <int F>
__global__ void k_selfinit(const float* __restrict__ h, const float* __restrict__ dinv,
                           const float* __restrict__ bias, float* __restrict__ out, int M) {
    long long t = (long long)blockIdx.x * blockDim.x + threadIdx.x;
    long long total = (long long)M * F;
    if (t >= total) return;
    int i = (int)(t / F);
    int f = (int)(t - (long long)i * F);
    float di = dinv[i];
    out[t] = h[t] * di * di + bias[f];
}

// out[dst] += h[src] * dinv[src]*dinv[dst], 4 features per thread
template <int F>
__global__ void k_scatter(const float* __restrict__ h, const long long* __restrict__ src,
                          const long long* __restrict__ dst, const float* __restrict__ dinv,
                          float* __restrict__ out, long long E) {
    constexpr int C = F / 4;
    long long t = (long long)blockIdx.x * blockDim.x + threadIdx.x;
    if (t >= E * C) return;
    long long e = t / C;
    int c = (int)(t - e * C);
    long long s = src[e], d = dst[e];
    float w = dinv[s] * dinv[d];
    const float4* hv = reinterpret_cast<const float4*>(h + (size_t)s * F);
    float4 v = hv[c];
    float* op = out + (size_t)d * F + 4 * c;
    atomicAdd(op + 0, v.x * w);
    atomicAdd(op + 1, v.y * w);
    atomicAdd(op + 2, v.z * w);
    atomicAdd(op + 3, v.w * w);
}

__global__ void k_relu_inplace(float* __restrict__ p, long long n) {
    long long i = (long long)blockIdx.x * blockDim.x + threadIdx.x;
    if (i < n) p[i] = fmaxf(p[i], 0.0f);
}

// ---------------------------------------------------------------------------
// bf16x3 WMMA GEMM:  C[M,NOUT] = A[M,K] @ W[K,NOUT]  (fp32 in/out, f32 accum)
// 256 threads = 8 wave32; block = 128 rows; each wave a 16-row strip.
// LDS tiles are stored PRE-SWIZZLED in WMMA fragment order, so each lane's
// fragment is one contiguous, 32B-aligned 16-element bf16 vector (2x b128).
//
// Fragment element mapping (ISA 7.12.2, 16-bit A 16x32):
//   k in [0,32):  half = (k>>3)&1 (lane group),  elem = ((k>>4)<<3) | (k&7)
//   A storage:  sA[(row*2 + half)*16 + elem]
//   B storage:  sB[((tile*32) + (col&15) + 16*half)*16 + elem]
// ---------------------------------------------------------------------------
template <int K, int NOUT>
__global__ void __launch_bounds__(256)
k_gemm_bf16x3(const float* __restrict__ A, const float* __restrict__ W,
              float* __restrict__ C, int M) {
    constexpr int KC = 32;
    constexpr int KP = ((K + KC - 1) / KC) * KC;
    constexpr int NT = NOUT / 16;

    __shared__ alignas(32) bf16_t sAhi[128 * KC];
    __shared__ alignas(32) bf16_t sAlo[128 * KC];
    __shared__ alignas(32) bf16_t sBhi[KC * NOUT];
    __shared__ alignas(32) bf16_t sBlo[KC * NOUT];

    const int tid  = threadIdx.x;
    const int lane = tid & 31;
    const int wave = tid >> 5;
    const int m    = lane & 15;   // column-in-tile / row-in-strip index
    const int rowBase = blockIdx.x * 128;

    const bf16x16* vAhi = reinterpret_cast<const bf16x16*>(sAhi);
    const bf16x16* vAlo = reinterpret_cast<const bf16x16*>(sAlo);
    const bf16x16* vBhi = reinterpret_cast<const bf16x16*>(sBhi);
    const bf16x16* vBlo = reinterpret_cast<const bf16x16*>(sBlo);

    floatx8 acc[NT];
#pragma unroll
    for (int t = 0; t < NT; ++t) acc[t] = (floatx8)0.0f;

    for (int kc = 0; kc < KP; kc += KC) {
        // Stage A tile (128 x KC) as bf16 hi/lo, swizzled, zero-padded.
#pragma unroll 4
        for (int idx = tid; idx < 128 * KC; idx += 256) {
            int r = idx / KC, c = idx - r * KC;
            int gr = rowBase + r, gk = kc + c;
            float v = (gr < M && gk < K) ? A[(size_t)gr * K + gk] : 0.0f;
            // prefetch next K-chunk of this row while we compute
            if ((c == 0 || c == 16) && gr < M && gk + KC < K)
                __builtin_prefetch(&A[(size_t)gr * K + gk + KC], 0, 0);
            bf16_t hi = (bf16_t)v;
            int sw = (r * 2 + ((c >> 3) & 1)) * 16 + (((c >> 4) << 3) | (c & 7));
            sAhi[sw] = hi;
            sAlo[sw] = (bf16_t)(v - (float)hi);
        }
        // Stage W chunk (KC x NOUT) as bf16 hi/lo, swizzled.
#pragma unroll 4
        for (int idx = tid; idx < KC * NOUT; idx += 256) {
            int kk = idx / NOUT, n = idx - kk * NOUT;
            int gk = kc + kk;
            float v = (gk < K) ? W[(size_t)gk * NOUT + n] : 0.0f;
            bf16_t hi = (bf16_t)v;
            int sw = (((n >> 4) << 5) + (n & 15) + 16 * ((kk >> 3) & 1)) * 16
                   + (((kk >> 4) << 3) | (kk & 7));
            sBhi[sw] = hi;
            sBlo[sw] = (bf16_t)(v - (float)hi);
        }
        __syncthreads();

        // A fragments: one contiguous 32B vector per lane.
        const int aIdx = (wave * 16 + m) * 2 + (lane >> 4);
        bf16x16 ahi = vAhi[aIdx];
        bf16x16 alo = vAlo[aIdx];

#pragma unroll
        for (int t = 0; t < NT; ++t) {
            bf16x16 bhi = vBhi[t * 32 + lane];
            bf16x16 blo = vBlo[t * 32 + lane];
            // bf16x3: hi*hi + hi*lo + lo*hi, f32 accumulate
            acc[t] = __builtin_amdgcn_wmma_f32_16x16x32_bf16(
                false, ahi, false, bhi, (short)0, acc[t], false, false);
            acc[t] = __builtin_amdgcn_wmma_f32_16x16x32_bf16(
                false, ahi, false, blo, (short)0, acc[t], false, false);
            acc[t] = __builtin_amdgcn_wmma_f32_16x16x32_bf16(
                false, alo, false, bhi, (short)0, acc[t], false, false);
        }
        __syncthreads();
    }

    // C/D layout: VGPR r -> M = r + 8*(lane>>4), column = lane&15.
#pragma unroll
    for (int t = 0; t < NT; ++t) {
#pragma unroll
        for (int r = 0; r < 8; ++r) {
            int row = rowBase + wave * 16 + r + 8 * (lane >> 4);
            if (row < M) C[(size_t)row * NOUT + t * 16 + m] = acc[t][r];
        }
    }
}

// ---------------------------------------------------------------------------
// Launch
// ---------------------------------------------------------------------------
static inline unsigned grid1d(long long work, int block) {
    return (unsigned)((work + block - 1) / block);
}

extern "C" void kernel_launch(void* const* d_in, const int* in_sizes, int n_in,
                              void* d_out, int out_size, void* d_ws, size_t ws_size,
                              hipStream_t stream) {
    (void)n_in; (void)out_size; (void)ws_size;
    const float* x  = (const float*)d_in[0];   // [N,100]
    const float* W1 = (const float*)d_in[1];   // [100,128]
    const float* b1 = (const float*)d_in[2];   // [128]
    const float* W2 = (const float*)d_in[3];   // [128,64]
    const float* b2 = (const float*)d_in[4];   // [64]
    const long long* ei = (const long long*)d_in[5];   // [2,E] int64

    const int N = in_sizes[0] / 100;
    const long long E = (long long)in_sizes[5] / 2;
    const long long* src = ei;
    const long long* dst = ei + E;

    // Workspace layout (floats): dinv[N] | tmp1[N*128] | agg1[N*128]
    float* dinv = (float*)d_ws;
    float* tmp1 = dinv + N;
    float* agg1 = tmp1 + (size_t)N * 128;
    float* tmp2 = tmp1;              // reuse after layer-1 scatter (N*64 <= N*128)
    float* outp = (float*)d_out;     // [N,64]

    const int B = 256;

    // --- degrees & normalization ---
    k_fill<<<grid1d(N, B), B, 0, stream>>>(dinv, 1.0f, N);                 // self loop
    k_deg_accum<<<grid1d(E, B), B, 0, stream>>>(dst, dinv, E);
    k_rsqrt_inplace<<<grid1d(N, B), B, 0, stream>>>(dinv, N);

    // --- layer 1: tmp1 = x @ W1 (WMMA bf16x3) ---
    k_gemm_bf16x3<100, 128><<<grid1d(N, 128), B, 0, stream>>>(x, W1, tmp1, N);
    // agg1 = self-loop + bias, then scatter messages, then ReLU
    k_selfinit<128><<<grid1d((long long)N * 128, B), B, 0, stream>>>(tmp1, dinv, b1, agg1, N);
    k_scatter<128><<<grid1d(E * 32, B), B, 0, stream>>>(tmp1, src, dst, dinv, agg1, E);
    k_relu_inplace<<<grid1d((long long)N * 128, B), B, 0, stream>>>(agg1, (long long)N * 128);

    // --- layer 2: tmp2 = h1 @ W2 (WMMA bf16x3) ---
    k_gemm_bf16x3<128, 64><<<grid1d(N, 128), B, 0, stream>>>(agg1, W2, tmp2, N);
    k_selfinit<64><<<grid1d((long long)N * 64, B), B, 0, stream>>>(tmp2, dinv, b2, outp, N);
    k_scatter<64><<<grid1d(E * 16, B), B, 0, stream>>>(tmp2, src, dst, dinv, outp, E);
}
